// SNN_83468394430669
// MI455X (gfx1250) — matrix operand order repaired and checked by reading, not compiled, not run on hardware
//
#include <hip/hip_runtime.h>
#include <hip/hip_bf16.h>

#define T_STEPS 1000
#define BATCH   128
#define FEAT    20
#define HID     256
#define OUTN    200
#define ALPHA_H 0.95f

typedef _Float16 half_t;
typedef __attribute__((ext_vector_type(16))) _Float16 v16h;
typedef __attribute__((ext_vector_type(8)))  _Float16 v8h;
typedef __attribute__((ext_vector_type(8)))  float    v8f;

static __device__ __forceinline__ v16h cat8(v8h lo, v8h hi) {
  return __builtin_shufflevector(lo, hi, 0,1,2,3,4,5,6,7,8,9,10,11,12,13,14,15);
}

// ---------------------------------------------------------------------------
// K0: weight prep. VrecT = transpose(Vrec) with zeroed diagonal (B-operand
// needs column-of-Vrec contiguous in K); W2/Wout rows are already K-contiguous.
// Wout padded to 256 rows of zeros so all 16 waves run full WMMA tiles.
// ---------------------------------------------------------------------------
__global__ void prep_weights(const float* __restrict__ Vrec1,
                             const float* __restrict__ Vrec2,
                             const float* __restrict__ W2,
                             const float* __restrict__ Wout,
                             half_t* __restrict__ vrec1T,
                             half_t* __restrict__ vrec2T,
                             half_t* __restrict__ w2h,
                             half_t* __restrict__ wouth) {
  const int g = blockIdx.x, h = threadIdx.x;
  vrec1T[g*HID + h] = (g == h) ? (half_t)0.f : (half_t)Vrec1[h*HID + g];
  vrec2T[g*HID + h] = (g == h) ? (half_t)0.f : (half_t)Vrec2[h*HID + g];
  w2h[g*HID + h]    = (half_t)W2[g*HID + h];
  wouth[g*HID + h]  = (g < OUTN) ? (half_t)Wout[g*HID + h] : (half_t)0.f;
}

// ---------------------------------------------------------------------------
// K1: Wx1[t*B+b, h] = b1[h] + sum_f x[b,t,f]*W1[h,f]   (K=20, memory bound)
// ---------------------------------------------------------------------------
__global__ __launch_bounds__(HID) void input_drive(const float* __restrict__ x,
                                                   const float* __restrict__ W1,
                                                   const float* __restrict__ b1,
                                                   float* __restrict__ Wx) {
  const int r = blockIdx.x;              // r = t*BATCH + b
  const int t = r / BATCH;
  const int b = r - t * BATCH;
  const int h = threadIdx.x;
  __shared__ float xs[FEAT];
  if (h < FEAT) xs[h] = x[((size_t)b*T_STEPS + t)*FEAT + h];
  __syncthreads();
  const float* wrow = W1 + h*FEAT;
  float acc = b1[h];
#pragma unroll
  for (int f = 0; f < FEAT; ++f) acc = fmaf(xs[f], wrow[f], acc);
  Wx[(size_t)r*HID + h] = acc;
}

// ---------------------------------------------------------------------------
// K2: recurrent scan. One block = 16 batch rows, 16 waves = 16 N-tiles.
// VrecT B-fragments stay resident in VGPRs for all 1000 steps. Wx[t+1] is
// software-pipelined into registers under the WMMA chain. The finished spike
// tile is drained LDS->global with global_store_async_from_lds_b128 (ASYNCcnt),
// overlapped with the next step's A-fragment loads.
// ---------------------------------------------------------------------------
__global__ __launch_bounds__(512) void recurrent_scan(const float* __restrict__ Wx,
                                                      const half_t* __restrict__ vrecT,
                                                      const float* __restrict__ beta,
                                                      half_t* __restrict__ spk_out) {
  __shared__ half_t tile[16*HID];              // 16 x 256 fp16 spike tile (8 KB)
  const int tid  = threadIdx.x;
  const int wave = tid >> 5;
  const int lane = tid & 31;
  const int hs   = lane >> 4;                  // half-wave select
  const int m    = lane & 15;                  // A-row / C-column index
  const int n    = wave*16 + m;                // output column owned by lane
  const int mb   = hs * 8;                     // C-fragment row base
  const int b0   = blockIdx.x * 16;

  // B operand: VrecT column slab, resident for the whole scan (8 x v16h).
  v16h bfrag[8];
#pragma unroll
  for (int kt = 0; kt < 8; ++kt) {
    const half_t* p = vrecT + (size_t)n*HID + kt*32 + hs*8;
    bfrag[kt] = cat8(*(const v8h*)p, *(const v8h*)(p + 16));
  }
  const float beta_n = beta[n];

  { v8h z = {0,0,0,0,0,0,0,0}; ((v8h*)tile)[tid] = z; }   // spk(t=-1) = 0
  __syncthreads();

  float syn[8] = {0,0,0,0,0,0,0,0};
  float mem[8] = {0,0,0,0,0,0,0,0};
  float spk[8] = {0,0,0,0,0,0,0,0};

  // Async-store bookkeeping: each thread drains 8 halves (16 B) of the tile.
  const unsigned lds_src   = (unsigned)(uintptr_t)(tile + tid*8);
  half_t* const  gdst_base = spk_out + (size_t)b0*HID + tid*8;

  // Pipeline prime: Wx[0]
  v8f wxc;
  {
    const float* wxp = Wx + ((size_t)b0 + mb)*HID + n;
#pragma unroll
    for (int v = 0; v < 8; ++v) wxc[v] = wxp[(size_t)v*HID];
  }

  for (int t = 0; t < T_STEPS; ++t) {
    // A fragments from previous step's spike tile
    v16h afrag[8];
#pragma unroll
    for (int kt = 0; kt < 8; ++kt) {
      const half_t* p = tile + m*HID + kt*32 + hs*8;
      afrag[kt] = cat8(*(const v8h*)p, *(const v8h*)(p + 16));
    }
    // our async drain of tile(t-1) must be done before anyone rewrites tile
    asm volatile("s_wait_asynccnt 0x0" ::: "memory");
    __syncthreads();                           // barrier 1

    v8f c = wxc;

    // prefetch Wx[t+1] into registers under the WMMA chain
    {
      const int tn = (t + 1 < T_STEPS) ? (t + 1) : t;
      const float* wxp = Wx + ((size_t)tn*BATCH + b0 + mb)*HID + n;
#pragma unroll
      for (int v = 0; v < 8; ++v) wxc[v] = wxp[(size_t)v*HID];
    }

#pragma unroll
    for (int kt = 0; kt < 8; ++kt)
      c = __builtin_amdgcn_wmma_f32_16x16x32_f16(false, afrag[kt],
                                                 false, bfrag[kt],
                                                 (short)0, c, false, false);

    // LIF update: syn = a*syn + act ; mem = b*mem + syn - spk_prev ; spike
#pragma unroll
    for (int v = 0; v < 8; ++v) {
      syn[v] = ALPHA_H * syn[v] + c[v];
      mem[v] = beta_n * mem[v] + syn[v] - spk[v];       // THR = 1
      float s = (mem[v] - 1.0f) > 0.0f ? 1.0f : 0.0f;
      spk[v] = s;
      tile[(mb + v)*HID + n] = (half_t)s;
    }
    __syncthreads();                           // barrier 2: tile(t) complete

    // async coalesced drain: tile(t) -> spk_out[t] (16 B per thread)
    {
      unsigned long long gaddr =
          (unsigned long long)(uintptr_t)(gdst_base + (size_t)t*BATCH*HID);
      asm volatile("global_store_async_from_lds_b128 %0, %1, off"
                   :: "v"(gaddr), "v"(lds_src) : "memory");
    }
  }
  asm volatile("s_wait_asynccnt 0x0" ::: "memory");
}

// ---------------------------------------------------------------------------
// K3: batched spike GEMM  out[r,n] = spk[r,:] @ Bw[n,:] (+bias[n])
// 32-row tiles (two 16x16 M-tiles share the per-wave B fragments -> half the
// B-matrix L2 traffic, and the two independent accumulators pipeline WMMAs).
// ---------------------------------------------------------------------------
__global__ __launch_bounds__(512) void spike_gemm(const half_t* __restrict__ A,
                                                  const half_t* __restrict__ Bw,
                                                  const float* __restrict__ bias,
                                                  float* __restrict__ out,
                                                  int nvalid, int ostride) {
  __shared__ half_t As[32*HID];                 // 16 KB
  const int tid  = threadIdx.x;
  const int wave = tid >> 5, lane = tid & 31;
  const int hs   = lane >> 4, m = lane & 15, mb = hs*8;
  const int n    = wave*16 + m;
  const size_t r0 = (size_t)blockIdx.x * 32;

  ((v8h*)As)[tid]       = ((const v8h*)(A + r0*HID))[tid];
  ((v8h*)As)[tid + 512] = ((const v8h*)(A + r0*HID))[tid + 512];
  __syncthreads();

  v8f c0, c1;
  const float bv = bias ? bias[n] : 0.f;
#pragma unroll
  for (int v = 0; v < 8; ++v) { c0[v] = bv; c1[v] = bv; }

#pragma unroll
  for (int kt = 0; kt < 8; ++kt) {
    const half_t* bp = Bw + (size_t)n*HID + kt*32 + hs*8;
    v16h b = cat8(*(const v8h*)bp, *(const v8h*)(bp + 16));
    const half_t* a0p = As + m*HID + kt*32 + hs*8;
    v16h a0 = cat8(*(const v8h*)a0p, *(const v8h*)(a0p + 16));
    const half_t* a1p = As + (16 + m)*HID + kt*32 + hs*8;
    v16h a1 = cat8(*(const v8h*)a1p, *(const v8h*)(a1p + 16));
    c0 = __builtin_amdgcn_wmma_f32_16x16x32_f16(false, a0, false, b,
                                                (short)0, c0, false, false);
    c1 = __builtin_amdgcn_wmma_f32_16x16x32_f16(false, a1, false, b,
                                                (short)0, c1, false, false);
  }

  if (n < nvalid) {
#pragma unroll
    for (int v = 0; v < 8; ++v) {
      out[(r0 + mb + v)*(size_t)ostride + n]      = c0[v];
      out[(r0 + 16 + mb + v)*(size_t)ostride + n] = c1[v];
    }
  }
}

// ---------------------------------------------------------------------------
// K4: readout Synaptic scan + softmax accumulation. One block per batch row.
// Wave32 shfl_xor reductions + two 8-entry LDS combines (4 barriers/step).
// ---------------------------------------------------------------------------
__global__ __launch_bounds__(256) void readout_scan(const float* __restrict__ outputs,
                                                    const float* __restrict__ alpha_o,
                                                    const float* __restrict__ beta_o,
                                                    float* __restrict__ dout) {
  const int b = blockIdx.x, o = threadIdx.x;
  const int lane = o & 31, wave = o >> 5;
  const bool active = (o < OUTN);
  __shared__ float red[8];
  float syn = 0.f, mem = 0.f, spk = 0.f, acc = 0.f;
  const float al = active ? alpha_o[o] : 0.f;
  const float be = active ? beta_o[o]  : 0.f;

  for (int t = 0; t < T_STEPS; ++t) {
    if (active) {
      float ot = outputs[((size_t)t*BATCH + b)*OUTN + o];
      syn = al*syn + ot;
      mem = be*mem + syn - spk;                 // THR = 1
      spk = (mem - 1.0f) > 0.0f ? 1.0f : 0.0f;
    }
    // block max via wave32 shuffles + 8-entry LDS combine
    float v = active ? mem : -3.0e38f;
#pragma unroll
    for (int off = 16; off > 0; off >>= 1) v = fmaxf(v, __shfl_xor(v, off));
    if (lane == 0) red[wave] = v;
    __syncthreads();
    float mx = red[0];
#pragma unroll
    for (int i = 1; i < 8; ++i) mx = fmaxf(mx, red[i]);
    __syncthreads();

    const float e = active ? __expf(mem - mx) : 0.f;
    float sv = e;
#pragma unroll
    for (int off = 16; off > 0; off >>= 1) sv += __shfl_xor(sv, off);
    if (lane == 0) red[wave] = sv;
    __syncthreads();
    float sum = 0.f;
#pragma unroll
    for (int i = 0; i < 8; ++i) sum += red[i];
    __syncthreads();

    if (t > 10 && active) acc += e / sum;       // WARMUP = 10
  }
  if (active) dout[(size_t)b*OUTN + o] = acc;
}

// ---------------------------------------------------------------------------
extern "C" void kernel_launch(void* const* d_in, const int* in_sizes, int n_in,
                              void* d_out, int out_size, void* d_ws, size_t ws_size,
                              hipStream_t stream) {
  const float* x         = (const float*)d_in[0];
  const float* W1        = (const float*)d_in[1];
  const float* b1        = (const float*)d_in[2];
  const float* Vrec1     = (const float*)d_in[3];
  const float* beta1     = (const float*)d_in[4];
  const float* W2        = (const float*)d_in[5];
  const float* b2        = (const float*)d_in[6];
  const float* Vrec2     = (const float*)d_in[7];
  const float* beta2     = (const float*)d_in[8];
  const float* Wout      = (const float*)d_in[9];
  const float* alpha_out = (const float*)d_in[10];
  const float* beta_out  = (const float*)d_in[11];
  float* out = (float*)d_out;

  char* ws = (char*)d_ws;
  const size_t szWx = (size_t)T_STEPS*BATCH*HID*sizeof(float);   // 131 MB (reused 3x)
  const size_t szSp = (size_t)T_STEPS*BATCH*HID*sizeof(half_t);  // 65.5 MB each
  const size_t szW  = (size_t)HID*HID*sizeof(half_t);
  float*  Wx     = (float*)(ws);
  half_t* spk1   = (half_t*)(ws + szWx);
  half_t* spk2   = (half_t*)(ws + szWx + szSp);
  half_t* vrec1T = (half_t*)(ws + szWx + 2*szSp);
  half_t* vrec2T = (half_t*)(ws + szWx + 2*szSp + szW);
  half_t* w2h    = (half_t*)(ws + szWx + 2*szSp + 2*szW);
  half_t* wouth  = (half_t*)(ws + szWx + 2*szSp + 3*szW);

  prep_weights<<<HID, HID, 0, stream>>>(Vrec1, Vrec2, W2, Wout,
                                        vrec1T, vrec2T, w2h, wouth);

  input_drive<<<T_STEPS*BATCH, HID, 0, stream>>>(x, W1, b1, Wx);

  recurrent_scan<<<BATCH/16, 512, 0, stream>>>(Wx, vrec1T, beta1, spk1);

  spike_gemm<<<T_STEPS*BATCH/32, 512, 0, stream>>>(spk1, w2h, b2, Wx, HID, HID);

  recurrent_scan<<<BATCH/16, 512, 0, stream>>>(Wx, vrec2T, beta2, spk2);

  spike_gemm<<<T_STEPS*BATCH/32, 512, 0, stream>>>(spk2, wouth, nullptr, Wx, OUTN, OUTN);

  readout_scan<<<BATCH, 256, 0, stream>>>(Wx, alpha_out, beta_out, out);
}
